// Encoder_87926570484537
// MI455X (gfx1250) — matrix operand order repaired and checked by reading, not compiled, not run on hardware
//
#include <hip/hip_runtime.h>

typedef __attribute__((ext_vector_type(16))) _Float16 v16h;
typedef __attribute__((ext_vector_type(8)))  _Float16 v8h;
typedef __attribute__((ext_vector_type(8)))  float    v8f;

#define N_NODES 50000
#define N_EDGES 800000
#define F_IN    128
#define DIM     500
#define N_TRIP  200000

#define NPAD    512                    // padded output width (32 N-tiles of 16)
#define MPAD    50048                  // 391 * 128, padded row count
#define KPAD2   512                    // padded K for GEMM2

union HV { v16h v16; v8h v8[2]; };
union U32H2 { unsigned u; _Float16 h[2]; };

// ---------------------------------------------------------------- utilities
__global__ void k_fill(float* __restrict__ p, float v, long n) {
    long i = (long)blockIdx.x * blockDim.x + threadIdx.x;
    long stride = (long)gridDim.x * blockDim.x;
    for (; i < n; i += stride) p[i] = v;
}

// zero-padded f32 -> f16 weight conversion into [Kpad x NPAD]
__global__ void k_padweights(const float* __restrict__ src, _Float16* __restrict__ dst,
                             int K, int Nout, long total) {
    long i = (long)blockIdx.x * blockDim.x + threadIdx.x;
    long stride = (long)gridDim.x * blockDim.x;
    for (; i < total; i += stride) {
        int k = (int)(i >> 9), n = (int)(i & (NPAD - 1));
        _Float16 v = (_Float16)0.0f;
        if (k < K && n < Nout) v = (_Float16)src[(long)k * Nout + n];
        dst[i] = v;
    }
}

__global__ void k_padbias(const float* __restrict__ src, float* __restrict__ dst, int Nout) {
    int n = blockIdx.x * blockDim.x + threadIdx.x;
    if (n < NPAD) dst[n] = (src != nullptr && n < Nout) ? src[n] : 0.0f;
}

// ---------------------------------------------------------------- degrees
__global__ void k_count_deg(const int* __restrict__ src, const int* __restrict__ dst,
                            float* __restrict__ dego, float* __restrict__ degi, int E) {
    int e = blockIdx.x * blockDim.x + threadIdx.x;
    if (e < E) {
        atomicAdd(&dego[src[e]], 1.0f);
        atomicAdd(&degi[dst[e]], 1.0f);
    }
}

__global__ void k_rsqrt_inplace(float* __restrict__ p, int n) {
    int i = blockIdx.x * blockDim.x + threadIdx.x;
    if (i < n) p[i] = rsqrtf(p[i]);   // deg >= 1 always (self-loop)
}

// ---------------------------------------------------------------- WMMA GEMM
// C[MPAD x NPAD] = A[rows x lda] @ Bh[Kpad x NPAD] + biasPad
// 8 waves/block; each wave computes one 16x32 C tile (two WMMA accumulators
// sharing the A fragment). B chunk staged transposed in LDS -> ds_load_b128.
// Branch-free inner loop: padding removes all bounds checks; B staging is
// explicitly unrolled (2 unconditional u32 loads per thread, no exec masks).
__global__ __launch_bounds__(256) void k_gemm_f16(
    const float* __restrict__ A, int lda, int rowClamp,
    const _Float16* __restrict__ Bh, const float* __restrict__ biasPad,
    float* __restrict__ C, int Kpad)
{
    __shared__ _Float16 Bs[32 * 32];            // transposed: Bs[n*32 + k]
    const int lane = threadIdx.x & 31;
    const int wave = threadIdx.x >> 5;
    const int hi   = lane >> 4;                 // lane half selects K sub-block
    const int col  = lane & 15;
    const int n0   = blockIdx.x * 32;
    const int m0   = (blockIdx.y * 8 + wave) * 16;

    int arow = m0 + col;
    if (arow > rowClamp) arow = rowClamp;       // branch-free clamp (stores stay padded)
    const float* Arow = A + (long)arow * lda;

    // per-thread staging coordinates: thread j-step p covers (k, n-pair)
    const int tid = threadIdx.x;

    v8f acc0 = {}, acc1 = {};
    for (int kc = 0; kc < Kpad; kc += 32) {
        // stage B chunk [32k x 32n] transposed into LDS.
        // step j: p = tid + 256*j ; k = p>>4 (0..31), n = (p&15)*2
        U32H2 cv0, cv1;
        {
            int p0 = tid;
            int k0 = p0 >> 4, nn0 = (p0 & 15) << 1;
            cv0.u = *(const unsigned*)(Bh + (long)(kc + k0) * NPAD + n0 + nn0);
            int p1 = tid + 256;
            int k1 = p1 >> 4, nn1 = (p1 & 15) << 1;
            cv1.u = *(const unsigned*)(Bh + (long)(kc + k1) * NPAD + n0 + nn1);
            Bs[(nn0    ) * 32 + k0] = cv0.h[0];
            Bs[(nn0 + 1) * 32 + k0] = cv0.h[1];
            Bs[(nn1    ) * 32 + k1] = cv1.h[0];
            Bs[(nn1 + 1) * 32 + k1] = cv1.h[1];
        }
        __syncthreads();

        if (kc + 32 < Kpad)
            __builtin_prefetch(Arow + kc + 32 + (hi << 3), 0, 1);

        // A fragment: K = (e&7) + 8*hi + 16*(e>=8)  -> two contiguous 8-float runs
        const float4* p0 = (const float4*)(Arow + kc + (hi << 3));
        const float4* p1 = (const float4*)(Arow + kc + 16 + (hi << 3));
        float4 r00 = p0[0], r01 = p0[1];
        float4 r10 = p1[0], r11 = p1[1];
        HV a;
        a.v16[0]  = (_Float16)r00.x; a.v16[1]  = (_Float16)r00.y;
        a.v16[2]  = (_Float16)r00.z; a.v16[3]  = (_Float16)r00.w;
        a.v16[4]  = (_Float16)r01.x; a.v16[5]  = (_Float16)r01.y;
        a.v16[6]  = (_Float16)r01.z; a.v16[7]  = (_Float16)r01.w;
        a.v16[8]  = (_Float16)r10.x; a.v16[9]  = (_Float16)r10.y;
        a.v16[10] = (_Float16)r10.z; a.v16[11] = (_Float16)r10.w;
        a.v16[12] = (_Float16)r11.x; a.v16[13] = (_Float16)r11.y;
        a.v16[14] = (_Float16)r11.z; a.v16[15] = (_Float16)r11.w;

        // B fragments: K = (e&7) + 8*hi + 16*(e>=8) in transposed LDS -> 2x ds_load_b128
        const v8h* bp0 = (const v8h*)(Bs + (col << 5) + (hi << 3));
        const v8h* bp1 = (const v8h*)(Bs + ((col + 16) << 5) + (hi << 3));
        HV b0, b1;
        b0.v8[0] = bp0[0]; b0.v8[1] = bp0[2];
        b1.v8[0] = bp1[0]; b1.v8[1] = bp1[2];

        acc0 = __builtin_amdgcn_wmma_f32_16x16x32_f16(
                   false, a.v16, false, b0.v16, (short)0, acc0, false, false);
        acc1 = __builtin_amdgcn_wmma_f32_16x16x32_f16(
                   false, a.v16, false, b1.v16, (short)0, acc1, false, false);
        __syncthreads();
    }

    // epilogue: C/D layout M = r + 8*hi, N = col; fully padded -> unguarded stores
    #pragma unroll
    for (int j = 0; j < 2; ++j) {
        int n = n0 + (j << 4) + col;
        float bv = biasPad[n];
        const v8f& acc = j ? acc1 : acc0;
        #pragma unroll
        for (int r = 0; r < 8; ++r) {
            long m = m0 + r + (hi << 3);
            C[m * NPAD + n] = acc[r] + bv;
        }
    }
}

// ---------------------------------------------------------------- GCN scatter
// agg[dst] += xw[src] * norm_src[src]; one block per (edge or self-loop)
__global__ __launch_bounds__(256) void k_scatter(
    const float* __restrict__ xw, const float* __restrict__ norms,
    const int* __restrict__ src, const int* __restrict__ dst,
    float* __restrict__ agg, int E)
{
    long e = blockIdx.x;
    int s, d;
    if (e < E) { s = src[e]; d = dst[e]; }
    else       { s = d = (int)(e - E); }      // self loop
    float ns = norms[s];
    const float* xr = xw + (long)s * NPAD;    // padded leading dim
    float* ar = agg + (long)d * DIM;          // output layout is tight
    for (int f = threadIdx.x; f < DIM; f += blockDim.x)
        atomicAdd(&ar[f], xr[f] * ns);
}

// embedding = agg * norm_dst + b_gcn    (in place, one block per node)
__global__ __launch_bounds__(256) void k_emb_final(
    float* __restrict__ emb, const float* __restrict__ normd,
    const float* __restrict__ bg)
{
    int node = blockIdx.x;
    float nv = normd[node];
    float* r = emb + (long)node * DIM;
    for (int f = threadIdx.x; f < DIM; f += blockDim.x)
        r[f] = r[f] * nv + bg[f];
}

// ---------------------------------------------------------------- predictor
// Collapse linear-linear:  v = W_in @ W_out  (1000),  c0 = b_in . W_out + b_out
__global__ void k_predvec(const float* __restrict__ Win, const float* __restrict__ bin,
                          const float* __restrict__ Wout, const float* __restrict__ bout,
                          float* __restrict__ v, float* __restrict__ c0)
{
    int k = blockIdx.x * blockDim.x + threadIdx.x;
    if (k < 2 * DIM) {
        float s = 0.0f;
        const float* row = Win + (long)k * DIM;
        for (int n = 0; n < DIM; ++n) s += row[n] * Wout[n];
        v[k] = s;
    } else if (k == 2 * DIM) {
        float s = bout[0];
        for (int n = 0; n < DIM; ++n) s += bin[n] * Wout[n];
        c0[0] = s;
    }
}

// weights[r] = emb[s].v[0:500] + emb[o].v[500:1000] + c0 ; one wave per triplet
__global__ __launch_bounds__(256) void k_weights(
    const float* __restrict__ emb, const float* __restrict__ v,
    const float* __restrict__ c0, const int* __restrict__ trip,
    float* __restrict__ w, int T)
{
    int lane = threadIdx.x & 31;
    int r = blockIdx.x * 8 + (threadIdx.x >> 5);
    if (r >= T) return;
    int s = trip[3L * r + 0];
    int o = trip[3L * r + 2];
    const float* es = emb + (long)s * DIM;
    const float* eo = emb + (long)o * DIM;
    float acc = 0.0f;
    for (int k = lane; k < DIM; k += 32)
        acc += es[k] * v[k] + eo[k] * v[DIM + k];
    #pragma unroll
    for (int off = 16; off > 0; off >>= 1)
        acc += __shfl_down(acc, off, 32);
    if (lane == 0) w[r] = acc + c0[0];
}

// ---------------------------------------------------------------- launcher
extern "C" void kernel_launch(void* const* d_in, const int* in_sizes, int n_in,
                              void* d_out, int out_size, void* d_ws, size_t ws_size,
                              hipStream_t stream) {
    const float* feats      = (const float*)d_in[0];
    const float* W_aff      = (const float*)d_in[1];
    const float* b_aff      = (const float*)d_in[2];
    const float* W_gcn      = (const float*)d_in[3];
    const float* b_gcn      = (const float*)d_in[4];
    const float* W_pred_in  = (const float*)d_in[5];
    const float* b_pred_in  = (const float*)d_in[6];
    const float* W_pred_out = (const float*)d_in[7];
    const float* b_pred_out = (const float*)d_in[8];
    const int*   src        = (const int*)d_in[9];
    const int*   dst        = (const int*)d_in[10];
    const int*   trip       = (const int*)d_in[11];

    float* weights = (float*)d_out;            // T elements
    float* emb     = weights + N_TRIP;         // N*DIM elements (also agg buffer)

    // workspace carve-up (padded buffers)
    float* x     = (float*)d_ws;                           // MPAD*NPAD
    float* xw    = x    + (size_t)MPAD * NPAD;             // MPAD*NPAD
    float* nrm_s = xw   + (size_t)MPAD * NPAD;             // N
    float* nrm_d = nrm_s + N_NODES;                        // N
    float* vvec  = nrm_d + N_NODES;                        // 2*DIM
    float* c0    = vvec + 2 * DIM;                         // 1
    float* biasA = c0 + 1;                                 // NPAD
    float* biasZ = biasA + NPAD;                           // NPAD
    _Float16* WhA = (_Float16*)(biasZ + NPAD);             // F_IN*NPAD
    _Float16* WhG = WhA + (size_t)F_IN * NPAD;             // KPAD2*NPAD

    // 1) padded f16 weights + padded biases (rewritten every call)
    k_padweights<<<256, 256, 0, stream>>>(W_aff, WhA, F_IN, DIM, (long)F_IN * NPAD);
    k_padweights<<<512, 256, 0, stream>>>(W_gcn, WhG, DIM, DIM, (long)KPAD2 * NPAD);
    k_padbias<<<2, 256, 0, stream>>>(b_aff, biasA, DIM);
    k_padbias<<<2, 256, 0, stream>>>(nullptr, biasZ, DIM);

    // 2) degrees (init 1.0 for self loop) -> rsqrt norms
    k_fill<<<256, 256, 0, stream>>>(nrm_s, 1.0f, N_NODES);
    k_fill<<<256, 256, 0, stream>>>(nrm_d, 1.0f, N_NODES);
    k_count_deg<<<(N_EDGES + 255) / 256, 256, 0, stream>>>(src, dst, nrm_s, nrm_d, N_EDGES);
    k_rsqrt_inplace<<<(N_NODES + 255) / 256, 256, 0, stream>>>(nrm_s, N_NODES);
    k_rsqrt_inplace<<<(N_NODES + 255) / 256, 256, 0, stream>>>(nrm_d, N_NODES);

    // 3) x = feats @ W_aff + b_aff   (K=128 exact multiple of 32)
    {
        dim3 grid(NPAD / 32, MPAD / 128);
        k_gemm_f16<<<grid, 256, 0, stream>>>(feats, F_IN, N_NODES - 1, WhA, biasA, x, F_IN);
    }
    // 4) xw = x @ W_gcn   (A padded: lda=NPAD, pad cols of x are exactly 0)
    {
        dim3 grid(NPAD / 32, MPAD / 128);
        k_gemm_f16<<<grid, 256, 0, stream>>>(x, NPAD, MPAD - 1, WhG, biasZ, xw, KPAD2);
    }

    // 5) agg (in d_out embedding region): zero, scatter, finalize
    k_fill<<<8192, 256, 0, stream>>>(emb, 0.0f, (long)N_NODES * DIM);
    k_scatter<<<N_EDGES + N_NODES, 256, 0, stream>>>(xw, nrm_s, src, dst, emb, N_EDGES);
    k_emb_final<<<N_NODES, 256, 0, stream>>>(emb, nrm_d, b_gcn);

    // 6) collapsed predictor: weights = concat(emb_s, emb_o) @ (Win@Wout) + (bin.Wout + bout)
    k_predvec<<<(2 * DIM + 1 + 255) / 256, 256, 0, stream>>>(
        W_pred_in, b_pred_in, W_pred_out, b_pred_out, vvec, c0);
    k_weights<<<(N_TRIP + 7) / 8, 256, 0, stream>>>(emb, vvec, c0, trip, weights, N_TRIP);
}
